// tree_attention_abstract_DP_32916629357288
// MI455X (gfx1250) — compile-verified, matrix operand order checked by reading, not run on hardware
//
#include <hip/hip_runtime.h>

// ============================================================================
// VQA forward for MI455X (gfx1250, wave32, WMMA f16 16x16x32, f32 accum).
// All big GEMMs (LSTM input proj, LSTM recurrence, conv1 3x3, residual 1x1 and
// 3x3 convs) run on v_wmma_f32_16x16x32_f16, software-pipelined so next-K
// fragment loads are in flight while current WMMAs execute (avoids
// s_wait_loadcnt 0 stalls).
// Coord channels are folded into precomputed per-(cout,pixel) terms so channel
// counts stay 32-aligned. Scratch requirement: ~230 MB in d_ws.
// ============================================================================

#define DEVI __device__ __forceinline__

typedef __attribute__((ext_vector_type(16))) _Float16 v16h;
typedef __attribute__((ext_vector_type(8)))  _Float16 v8h;
typedef __attribute__((ext_vector_type(8)))  float    v8f;

// ---------------------------------------------------------------------------
// WMMA helpers. 16-bit A-matrix 16x32 layout (ISA 7.12.2): lane = g*16 + r,
// lane holds row r, K = {g*8..g*8+7} in VGPR0..3 and K = {16+g*8..16+g*8+7}
// in VGPR4..7 -> two contiguous 8-half (16B) chunks. B sourced from W[N,K]
// row-major uses the identical per-lane pattern with r = column n.
// ---------------------------------------------------------------------------
DEVI v16h frag_ld(const _Float16* __restrict__ rowbase, int g) {
  v8h a0 = *(const v8h*)(rowbase + g * 8);
  v8h a1 = *(const v8h*)(rowbase + 16 + g * 8);
  v16h r;
#pragma unroll
  for (int i = 0; i < 8; ++i) { r[i] = a0[i]; r[i + 8] = a1[i]; }
  return r;
}

DEVI v8f wmma_f16(v16h a, v16h b, v8f c) {
  return __builtin_amdgcn_wmma_f32_16x16x32_f16(
      /*neg_a=*/false, a, /*neg_b=*/false, b,
      /*c_mod=*/(short)0, c, /*reuse_a=*/false, /*reuse_b=*/false);
}

DEVI float sigm(float x) { return 1.f / (1.f + __expf(-x)); }

// ---------------------------------------------------------------------------
// Small utility kernels
// ---------------------------------------------------------------------------
__global__ void k_fill16(_Float16* p, long n) {
  long i = (long)blockIdx.x * 256 + threadIdx.x;
  if (i < n) p[i] = (_Float16)0.f;
}
__global__ void k_fill32(float* p, long n) {
  long i = (long)blockIdx.x * 256 + threadIdx.x;
  if (i < n) p[i] = 0.f;
}
__global__ void k_cast_f16(const float* __restrict__ s, _Float16* __restrict__ d, long n) {
  long i = (long)blockIdx.x * 256 + threadIdx.x;
  if (i < n) d[i] = (_Float16)s[i];
}
__global__ void k_bias_sum(const float* a, const float* b, float* o, int n) {
  int i = blockIdx.x * 256 + threadIdx.x;
  if (i < n) o[i] = a[i] + b[i];
}

// Embedding lookup + cast, x16[s][b][0..320) (300 real + 20 zero pad), padding_idx=0.
__global__ void k_embed(const int* __restrict__ que, const float* __restrict__ emb,
                        _Float16* __restrict__ x16) {
  long i = (long)blockIdx.x * 256 + threadIdx.x;
  if (i >= 4096L * 320) return;
  int k = (int)(i % 320);
  int sb = (int)(i / 320);
  int s = sb / 128, b = sb % 128;
  int tok = que[b * 32 + s];
  float v = (k < 300 && tok != 0) ? emb[(long)tok * 300 + k] : 0.f;
  x16[i] = (_Float16)v;
}

// W_ih [4096,300] -> f16 [4096,320] zero-padded in K.
__global__ void k_pack_wih(const float* __restrict__ w, _Float16* __restrict__ wp) {
  long i = (long)blockIdx.x * 256 + threadIdx.x;
  if (i >= 4096L * 320) return;
  int k = (int)(i % 320);
  long n = i / 320;
  wp[i] = (_Float16)((k < 300) ? w[n * 300 + k] : 0.f);
}

// ---------------------------------------------------------------------------
// Generic GEMM: C[M,N] = A[M,K](f16,row-major) * W[N,K](f16,row-major)^T + bias
// block = 128 thr (4 waves); block tile 64M x 64N; wave tile 16M x 64N.
// Software-pipelined: fragments for k+32 load while k's WMMAs execute.
// ---------------------------------------------------------------------------
__global__ void k_gemm_bias(const _Float16* __restrict__ A, const _Float16* __restrict__ W,
                            const float* __restrict__ bias, float* __restrict__ C,
                            int M, int N, int K) {
  int wave = threadIdx.x >> 5, lane = threadIdx.x & 31;
  int g = lane >> 4, r = lane & 15;
  int m0 = blockIdx.x * 64 + wave * 16;
  int n0 = blockIdx.y * 64;
  const _Float16* arow = A + (long)(m0 + r) * K;
  const _Float16* wrow = W + (long)(n0 + r) * K;  // fragment j adds j*16*K
  v8f acc[4] = {};
  v16h a = frag_ld(arow, g);
  v16h b[4];
#pragma unroll
  for (int j = 0; j < 4; ++j) b[j] = frag_ld(wrow + (long)j * 16 * K, g);
  for (int k = 32; k < K; k += 32) {
    __builtin_prefetch(arow + k + 96, 0, 3);
    v16h an = frag_ld(arow + k, g);
    v16h bn[4];
#pragma unroll
    for (int j = 0; j < 4; ++j) bn[j] = frag_ld(wrow + (long)j * 16 * K + k, g);
#pragma unroll
    for (int j = 0; j < 4; ++j) acc[j] = wmma_f16(a, b[j], acc[j]);
    a = an;
#pragma unroll
    for (int j = 0; j < 4; ++j) b[j] = bn[j];
  }
#pragma unroll
  for (int j = 0; j < 4; ++j) acc[j] = wmma_f16(a, b[j], acc[j]);

#pragma unroll
  for (int j = 0; j < 4; ++j) {
    int n = n0 + j * 16 + r;
    float bv = bias ? bias[n] : 0.f;
#pragma unroll
    for (int v = 0; v < 8; ++v) {
      int m = m0 + v + 8 * g;  // C/D layout: m = vgpr + 8*lanegroup, n = lane&15
      C[(long)m * N + n] = acc[j][v] + bv;
    }
  }
}

// ---------------------------------------------------------------------------
// Fused LSTM step: gates = h_prev @ W_hh^T + gates_in[s]; cell update.
// Wave computes 16(B) x 16(J) for all 4 gates (4 accumulators share one A frag).
// grid (8, 16), block 128. Software-pipelined K loop.
// ---------------------------------------------------------------------------
__global__ void k_lstm_step(const float* __restrict__ gates_in,  // [S*B,4096] (incl. biases)
                            const _Float16* __restrict__ Whh,    // [4096,1024]
                            const _Float16* __restrict__ hprev,  // [128,1024] f16
                            float* __restrict__ c,               // [128,1024] (in-place)
                            float* __restrict__ h_all,           // [32,128,1024]
                            _Float16* __restrict__ hnext,        // [128,1024] f16
                            int s) {
  int wave = threadIdx.x >> 5, lane = threadIdx.x & 31;
  int g = lane >> 4, r = lane & 15;
  int m0 = blockIdx.x * 16;
  int j0 = blockIdx.y * 64 + wave * 16;
  const _Float16* arow = hprev + (long)(m0 + r) * 1024;
  const _Float16* wrow = Whh + (long)(j0 + r) * 1024;  // gate q adds q*1024*1024
  v8f acc[4] = {};
  v16h a = frag_ld(arow, g);
  v16h b[4];
#pragma unroll
  for (int q = 0; q < 4; ++q) b[q] = frag_ld(wrow + (long)q * 1024 * 1024, g);
  for (int k = 32; k < 1024; k += 32) {
    __builtin_prefetch(arow + k + 96, 0, 3);
    v16h an = frag_ld(arow + k, g);
    v16h bn[4];
#pragma unroll
    for (int q = 0; q < 4; ++q) bn[q] = frag_ld(wrow + (long)q * 1024 * 1024 + k, g);
#pragma unroll
    for (int q = 0; q < 4; ++q) acc[q] = wmma_f16(a, b[q], acc[q]);
    a = an;
#pragma unroll
    for (int q = 0; q < 4; ++q) b[q] = bn[q];
  }
#pragma unroll
  for (int q = 0; q < 4; ++q) acc[q] = wmma_f16(a, b[q], acc[q]);

#pragma unroll
  for (int v = 0; v < 8; ++v) {
    int m = m0 + v + 8 * g;  // batch row
    int j = j0 + r;          // hidden unit
    long grow = ((long)s * 128 + m) * 4096;
    float gi = acc[0][v] + gates_in[grow + j];
    float gf = acc[1][v] + gates_in[grow + 1024 + j];
    float gg = acc[2][v] + gates_in[grow + 2048 + j];
    float go = acc[3][v] + gates_in[grow + 3072 + j];
    long idx = (long)m * 1024 + j;
    float cn = sigm(gf) * c[idx] + sigm(gi) * tanhf(gg);
    c[idx] = cn;
    float h = sigm(go) * tanhf(cn);
    h_all[((long)s * 128 + m) * 1024 + j] = h;
    hnext[idx] = (_Float16)h;
  }
}

// qenc L2-normalize: one block per (s,b); also writes enc for s==31.
__global__ void k_qenc(const float* __restrict__ hf, const float* __restrict__ hb,
                       float* __restrict__ out_enc, float* __restrict__ out_qenc) {
  int sb = blockIdx.x;
  int s = sb / 128, b = sb % 128;
  const float* pf = hf + (long)sb * 1024;
  const float* pb = hb + (long)sb * 1024;
  __shared__ float red[256];
  float ss = 0.f;
  for (int k = threadIdx.x; k < 1024; k += 256) {
    float a = pf[k], c = pb[k];
    ss += a * a + c * c;
  }
  red[threadIdx.x] = ss; __syncthreads();
  for (int st = 128; st > 0; st >>= 1) {
    if (threadIdx.x < st) red[threadIdx.x] += red[threadIdx.x + st];
    __syncthreads();
  }
  float sc = 1.f / fmaxf(sqrtf(red[0]), 1e-12f);
  float* q = out_qenc + (long)sb * 2048;
  for (int k = threadIdx.x; k < 1024; k += 256) { q[k] = pf[k] * sc; q[1024 + k] = pb[k] * sc; }
  if (s == 31) {
    float* e = out_enc + (long)b * 2048;
    for (int k = threadIdx.x; k < 1024; k += 256) { e[k] = pf[k] * sc; e[1024 + k] = pb[k] * sc; }
  }
}

// ---------------------------------------------------------------------------
// Image branch
// ---------------------------------------------------------------------------
// Per-pixel channel L2 norm; output NHWC f16 with 1-px zero border: [B,16,16,1024].
__global__ void k_l2norm_img(const float* __restrict__ img, _Float16* __restrict__ vp) {
  int pix = blockIdx.x;
  int b = pix / 196, p = pix % 196, y = p / 14, x = p % 14;
  __shared__ float red[256];
  float ss = 0.f;
  for (int c = threadIdx.x; c < 1024; c += 256) {
    float t = img[((long)b * 1024 + c) * 196 + p];
    ss += t * t;
  }
  red[threadIdx.x] = ss; __syncthreads();
  for (int st = 128; st > 0; st >>= 1) {
    if (threadIdx.x < st) red[threadIdx.x] += red[threadIdx.x + st];
    __syncthreads();
  }
  float sc = 1.f / fmaxf(sqrtf(red[0]), 1e-12f);
  long base = (((long)b * 16 + y + 1) * 16 + x + 1) * 1024;
  for (int c = threadIdx.x; c < 1024; c += 256)
    vp[base + c] = (_Float16)(img[((long)b * 1024 + c) * 196 + p] * sc);
}

DEVI float coordv(int ch, int pp) {  // pp = y*14+x flat index (matches reference _coord)
  return (ch == 0) ? ((float)pp / 14.f - 7.f) / 7.f : ((float)(pp % 14) - 7.f) / 7.f;
}

// Coord-channel contribution for the 3x3 conv1 (w OIHW [128,Ci,3,3], coords = last 2 ch).
__global__ void k_coordterm3(const float* __restrict__ w, const float* __restrict__ bias,
                             float* __restrict__ ct, int Ci) {
  int i = blockIdx.x * 256 + threadIdx.x;
  if (i >= 128 * 196) return;
  int p = i % 196, co = i / 196, y = p / 14, x = p % 14;
  float acc = bias[co];
  for (int c = 0; c < 2; ++c)
    for (int dy = 0; dy < 3; ++dy)
      for (int dx = 0; dx < 3; ++dx) {
        int iy = y + dy - 1, ix = x + dx - 1;
        if (iy < 0 || iy > 13 || ix < 0 || ix > 13) continue;
        acc += w[(((long)co * Ci + (Ci - 2 + c)) * 3 + dy) * 3 + dx] * coordv(c, iy * 14 + ix);
      }
  ct[i] = acc;
}

// Coord-channel contribution for residual 1x1 (w [128,130,1,1]).
__global__ void k_coordterm1(const float* __restrict__ w, const float* __restrict__ bias,
                             float* __restrict__ ct) {
  int i = blockIdx.x * 256 + threadIdx.x;
  if (i >= 128 * 196) return;
  int p = i % 196, co = i / 196;
  ct[i] = bias[co] + w[co * 130 + 128] * coordv(0, p) + w[co * 130 + 129] * coordv(1, p);
}

__global__ void k_bias_tile(const float* __restrict__ bias, float* __restrict__ ct) {
  int i = blockIdx.x * 256 + threadIdx.x;
  if (i >= 128 * 196) return;
  ct[i] = bias[i / 196];
}

// OIHW [128,CiSrc,3,3] -> f16 [128][tap=9][CiUse] (tap-major K for implicit gemm).
__global__ void k_pack_conv3(const float* __restrict__ w, _Float16* __restrict__ wp,
                             int CiSrc, int CiUse) {
  long i = (long)blockIdx.x * 256 + threadIdx.x;
  long n = (long)128 * 9 * CiUse;
  if (i >= n) return;
  int c = (int)(i % CiUse);
  int tap = (int)((i / CiUse) % 9);
  int co = (int)(i / ((long)9 * CiUse));
  wp[i] = (_Float16)w[(((long)co * CiSrc + c) * 3 + tap / 3) * 3 + (tap % 3)];
}

__global__ void k_pack_conv1(const float* __restrict__ w, _Float16* __restrict__ wp) {
  int i = blockIdx.x * 256 + threadIdx.x;
  if (i >= 128 * 128) return;
  wp[i] = (_Float16)w[(i / 128) * 130 + (i % 128)];
}

// ---------------------------------------------------------------------------
// Implicit-GEMM conv. A rows = output pixels (b,oy,ox), K = taps x Ci
// (tap-major, so the packed-W byte offset for flat K-step t is simply t*32
// halves). Input NHWC f16 with zero border [B,16,16,Ci]. Wave tile 16M x 64N
// (4 accumulators), grid (392, 128/64), block 128 (4 waves). Software-
// pipelined across the flattened tap*chunk loop. Epilogue adds
// cterm[cout*196+p] (coord part + bias), optional relu, writes f32 NHWC
// and/or f16 zero-bordered NHWC.
// ---------------------------------------------------------------------------
__global__ void k_conv_gemm(const _Float16* __restrict__ in, const _Float16* __restrict__ Wp,
                            const float* __restrict__ cterm, float* __restrict__ outF32,
                            _Float16* __restrict__ outF16pad, int Ci, int taps, int relu) {
  int wave = threadIdx.x >> 5, lane = threadIdx.x & 31;
  int g = lane >> 4, r = lane & 15;
  int mt = blockIdx.x * 4 + wave;  // 0..1567 M-tiles of 16 pixels
  int n0 = blockIdx.y * 64;
  int pix = mt * 16 + r;
  int b = pix / 196, p = pix % 196, oy = p / 14, ox = p % 14;
  int d0 = (taps == 1) ? 1 : 0;  // 1x1 conv samples the padded center
  const _Float16* rowbase = in + (((long)b * 16 + oy + d0) * 16 + ox + d0) * Ci;
  long ldw = (long)taps * Ci;
  const _Float16* wrow = Wp + (long)(n0 + r) * ldw;  // fragment j adds j*16*ldw
  int KC = Ci >> 5;
  int nk = taps * KC;
  v8f acc[4] = {};

  int tap = 0, kc = 0;
  const _Float16* abase = rowbase;
  v16h a = frag_ld(abase, g);
  v16h bfr[4];
#pragma unroll
  for (int j = 0; j < 4; ++j) bfr[j] = frag_ld(wrow + (long)j * 16 * ldw, g);
  for (int t = 1; t < nk; ++t) {
    ++kc;
    if (kc == KC) { kc = 0; ++tap; abase = rowbase + ((tap / 3) * 16 + (tap % 3)) * Ci; }
    __builtin_prefetch(abase + kc * 32 + 96, 0, 3);
    v16h an = frag_ld(abase + kc * 32, g);
    v16h bn[4];
#pragma unroll
    for (int j = 0; j < 4; ++j) bn[j] = frag_ld(wrow + (long)j * 16 * ldw + (long)t * 32, g);
#pragma unroll
    for (int j = 0; j < 4; ++j) acc[j] = wmma_f16(a, bfr[j], acc[j]);
    a = an;
#pragma unroll
    for (int j = 0; j < 4; ++j) bfr[j] = bn[j];
  }
#pragma unroll
  for (int j = 0; j < 4; ++j) acc[j] = wmma_f16(a, bfr[j], acc[j]);

#pragma unroll
  for (int j = 0; j < 4; ++j) {
    int n = n0 + j * 16 + r;  // cout
#pragma unroll
    for (int v = 0; v < 8; ++v) {
      int m = mt * 16 + v + 8 * g;
      int bb = m / 196, pp = m % 196;
      float val = acc[j][v] + cterm[n * 196 + pp];
      if (relu) val = fmaxf(val, 0.f);
      if (outF32) outF32[(long)m * 128 + n] = val;
      if (outF16pad) {
        int yy = pp / 14, xx = pp % 14;
        outF16pad[(((long)bb * 16 + yy + 1) * 16 + xx + 1) * 128 + n] = (_Float16)val;
      }
    }
  }
}

// BatchNorm batch statistics per channel over NHWC f32 [25088,128].
__global__ void k_bn_stats(const float* __restrict__ y, float* __restrict__ stats) {
  int c = blockIdx.x;
  __shared__ float rs[256], rq[256];
  float s = 0.f, q = 0.f;
  for (int i = threadIdx.x; i < 25088; i += 256) {
    float t = y[(long)i * 128 + c];
    s += t; q += t * t;
  }
  rs[threadIdx.x] = s; rq[threadIdx.x] = q; __syncthreads();
  for (int st = 128; st > 0; st >>= 1) {
    if (threadIdx.x < st) { rs[threadIdx.x] += rs[threadIdx.x + st]; rq[threadIdx.x] += rq[threadIdx.x + st]; }
    __syncthreads();
  }
  if (threadIdx.x == 0) {
    float m = rs[0] / 25088.f;
    stats[c] = m;
    stats[128 + c] = rq[0] / 25088.f - m * m;
  }
}

// BN + ReLU (+ optional residual add of v1). mode 0: ->f16pad. mode 1: +v1 ->f16pad.
// mode 2: +v1 -> f32 NCHW (final output).
__global__ void k_bn_apply(const float* __restrict__ y, const float* __restrict__ stats,
                           const float* __restrict__ gamma, const float* __restrict__ beta,
                           const _Float16* __restrict__ v1pad, _Float16* __restrict__ outPad,
                           float* __restrict__ outNCHW, int mode) {
  long i = (long)blockIdx.x * 256 + threadIdx.x;
  if (i >= 25088L * 128) return;
  int c = (int)(i & 127);
  int pixp = (int)(i >> 7);
  int b = pixp / 196, p = pixp % 196, yy = p / 14, xx = p % 14;
  float m = stats[c], var = stats[128 + c];
  float val = gamma[c] * (y[i] - m) * rsqrtf(var + 1e-5f) + beta[c];
  val = fmaxf(val, 0.f);
  long padidx = (((long)b * 16 + yy + 1) * 16 + xx + 1) * 128 + c;
  if (mode != 0) val += (float)v1pad[padidx];
  if (mode == 2) outNCHW[((long)b * 128 + c) * 196 + p] = val;
  else outPad[padidx] = (_Float16)val;
}

// ---------------------------------------------------------------------------
// Host launch
// ---------------------------------------------------------------------------
static inline int cdiv(long n, int d) { return (int)((n + d - 1) / d); }

extern "C" void kernel_launch(void* const* d_in, const int* in_sizes, int n_in,
                              void* d_out, int out_size, void* d_ws, size_t ws_size,
                              hipStream_t stream) {
  (void)in_sizes; (void)n_in; (void)out_size; (void)ws_size;
  const int*   que  = (const int*)d_in[0];
  const float* img  = (const float*)d_in[1];
  const float* emb  = (const float*)d_in[2];
  const float* w_ih[2] = {(const float*)d_in[3], (const float*)d_in[7]};
  const float* w_hh[2] = {(const float*)d_in[4], (const float*)d_in[8]};
  const float* b_ih[2] = {(const float*)d_in[5], (const float*)d_in[9]};
  const float* b_hh[2] = {(const float*)d_in[6], (const float*)d_in[10]};
  const float* conv_w = (const float*)d_in[11];
  const float* conv_b = (const float*)d_in[12];
  const float* bn_g   = (const float*)d_in[13];
  const float* bn_b   = (const float*)d_in[14];
  const float* rc1_w[2] = {(const float*)d_in[15], (const float*)d_in[21]};
  const float* rc1_b[2] = {(const float*)d_in[16], (const float*)d_in[22]};
  const float* rc2_w[2] = {(const float*)d_in[17], (const float*)d_in[23]};
  const float* rc2_b[2] = {(const float*)d_in[18], (const float*)d_in[24]};
  const float* rbn_g[2] = {(const float*)d_in[19], (const float*)d_in[25]};
  const float* rbn_b[2] = {(const float*)d_in[20], (const float*)d_in[26]};

  float* out      = (float*)d_out;
  float* out_enc  = out;                 // [128,2048]
  float* out_qenc = out + 262144;        // [32,128,2048]
  float* out_v    = out + 8650752;       // [128,128,14,14]

  // ---- workspace carve (256B aligned) ----
  size_t off = 0;
  char* base = (char*)d_ws;
  auto carve = [&](size_t bytes) -> char* {
    char* p = base + off;
    off += (bytes + 255) & ~(size_t)255;
    return p;
  };
  _Float16* x16      = (_Float16*)carve(4096L * 320 * 2);
  _Float16* wih16[2] = {(_Float16*)carve(4096L * 320 * 2), (_Float16*)carve(4096L * 320 * 2)};
  _Float16* whh16[2] = {(_Float16*)carve(4096L * 1024 * 2), (_Float16*)carve(4096L * 1024 * 2)};
  float*    biasg[2] = {(float*)carve(4096 * 4), (float*)carve(4096 * 4)};
  float*    gates    = (float*)carve(4096L * 4096 * 4);           // reused per direction
  float*    hall[2]  = {(float*)carve(32L * 128 * 1024 * 4), (float*)carve(32L * 128 * 1024 * 4)};
  _Float16* hp[2]    = {(_Float16*)carve(128L * 1024 * 2), (_Float16*)carve(128L * 1024 * 2)};
  float*    cbuf     = (float*)carve(128L * 1024 * 4);
  _Float16* vnorm    = (_Float16*)carve(128L * 16 * 16 * 1024 * 2);
  _Float16* w1p      = (_Float16*)carve(128L * 9 * 1024 * 2);
  float*    ct       = (float*)carve(128L * 196 * 4);
  float*    ybuf     = (float*)carve(25088L * 128 * 4);
  float*    stats    = (float*)carve(256 * 4);
  _Float16* vA       = (_Float16*)carve(128L * 16 * 16 * 128 * 2);
  _Float16* v1p      = (_Float16*)carve(128L * 16 * 16 * 128 * 2);
  _Float16* wc1p     = (_Float16*)carve(128L * 128 * 2);
  _Float16* wc2p     = (_Float16*)carve(128L * 9 * 128 * 2);

  // ================= Text branch =================
  k_embed<<<cdiv(4096L * 320, 256), 256, 0, stream>>>(que, emb, x16);
  for (int d = 0; d < 2; ++d) {
    k_pack_wih<<<cdiv(4096L * 320, 256), 256, 0, stream>>>(w_ih[d], wih16[d]);
    k_cast_f16<<<cdiv(4096L * 1024, 256), 256, 0, stream>>>(w_hh[d], whh16[d], 4096L * 1024);
    k_bias_sum<<<16, 256, 0, stream>>>(b_ih[d], b_hh[d], biasg[d], 4096);
  }
  for (int d = 0; d < 2; ++d) {
    // gates_in = X @ W_ih^T + (b_ih + b_hh), M=4096, N=4096, K=320
    k_gemm_bias<<<dim3(64, 64), 128, 0, stream>>>(x16, wih16[d], biasg[d], gates, 4096, 4096, 320);
    k_fill16<<<cdiv(128L * 1024, 256), 256, 0, stream>>>(hp[0], 128L * 1024);
    k_fill16<<<cdiv(128L * 1024, 256), 256, 0, stream>>>(hp[1], 128L * 1024);
    k_fill32<<<cdiv(128L * 1024, 256), 256, 0, stream>>>(cbuf, 128L * 1024);
    for (int t = 0; t < 32; ++t) {
      int s = d ? (31 - t) : t;  // backward scan consumes x[S-1-t], stores h at s
      k_lstm_step<<<dim3(8, 16), 128, 0, stream>>>(gates, whh16[d], hp[t & 1], cbuf,
                                                   hall[d], hp[(t + 1) & 1], s);
    }
  }
  k_qenc<<<4096, 256, 0, stream>>>(hall[0], hall[1], out_enc, out_qenc);

  // ================= Image branch =================
  k_fill16<<<cdiv(128L * 16 * 16 * 1024, 256), 256, 0, stream>>>(vnorm, 128L * 16 * 16 * 1024);
  k_fill16<<<cdiv(128L * 16 * 16 * 128, 256), 256, 0, stream>>>(vA, 128L * 16 * 16 * 128);
  k_fill16<<<cdiv(128L * 16 * 16 * 128, 256), 256, 0, stream>>>(v1p, 128L * 16 * 16 * 128);
  k_l2norm_img<<<25088, 256, 0, stream>>>(img, vnorm);

  // conv1: 3x3 over 1024 f16 channels (+ coord term), then BN(batch stats)+ReLU
  k_coordterm3<<<98, 256, 0, stream>>>(conv_w, conv_b, ct, 1026);
  k_pack_conv3<<<cdiv(128L * 9 * 1024, 256), 256, 0, stream>>>(conv_w, w1p, 1026, 1024);
  k_conv_gemm<<<dim3(392, 2), 128, 0, stream>>>(vnorm, w1p, ct, ybuf, nullptr, 1024, 9, 0);
  k_bn_stats<<<128, 256, 0, stream>>>(ybuf, stats);
  k_bn_apply<<<cdiv(25088L * 128, 256), 256, 0, stream>>>(ybuf, stats, bn_g, bn_b,
                                                          nullptr, vA, nullptr, 0);

  // residual blocks
  for (int rblk = 0; rblk < 2; ++rblk) {
    k_coordterm1<<<98, 256, 0, stream>>>(rc1_w[rblk], rc1_b[rblk], ct);
    k_pack_conv1<<<64, 256, 0, stream>>>(rc1_w[rblk], wc1p);
    // v1 = relu(1x1(v) + coord term) -> f16 padded
    k_conv_gemm<<<dim3(392, 2), 128, 0, stream>>>(vA, wc1p, ct, nullptr, v1p, 128, 1, 1);
    k_bias_tile<<<98, 256, 0, stream>>>(rc2_b[rblk], ct);
    k_pack_conv3<<<cdiv(128L * 9 * 128, 256), 256, 0, stream>>>(rc2_w[rblk], wc2p, 128, 128);
    // y2 = 3x3(v1) + b
    k_conv_gemm<<<dim3(392, 2), 128, 0, stream>>>(v1p, wc2p, ct, ybuf, nullptr, 128, 9, 0);
    k_bn_stats<<<128, 256, 0, stream>>>(ybuf, stats);
    // out = relu(bn(y2)) + v1 ; block0 -> vA (f16 padded), block1 -> final NCHW f32
    k_bn_apply<<<cdiv(25088L * 128, 256), 256, 0, stream>>>(ybuf, stats, rbn_g[rblk], rbn_b[rblk],
                                                            v1p, vA, out_v, (rblk == 0) ? 1 : 2);
  }
}
